// MLA_8469675508070
// MI455X (gfx1250) — matrix operand order repaired and checked by reading, not compiled, not run on hardware
//
#include <hip/hip_runtime.h>
#include <hip/hip_bf16.h>
#include <math.h>

#define B_      2
#define S_      2048
#define D_      2048
#define H_      16
#define QLORA_  1024
#define KVLORA_ 512
#define NOPE_   128
#define ROPE_   64
#define VH_     128
#define QK_     192
#define CAT_    576      /* KVLORA + ROPE */
#define CATP_   584      /* padded LDS row (16B aligned, bank-spread) */
#define EPS_    1e-3f

typedef __attribute__((ext_vector_type(16))) __bf16   v16bf;
typedef __attribute__((ext_vector_type(8)))  float    v8f;
typedef __attribute__((ext_vector_type(4)))  unsigned v4u;

union Frag16 { v16bf v; __bf16 h[16]; unsigned short u[16]; v4u u4[2]; };

__device__ __forceinline__ v8f wmma_bf16(v16bf a, v16bf b, v8f c) {
  return __builtin_amdgcn_wmma_f32_16x16x32_bf16(false, a, false, b, (short)0, c,
                                                 false, false);
}

// Async 16-byte global -> LDS copy (CDNA5 GLOBAL_LOAD_ASYNC_TO_LDS_B128).
// Per-lane: copies 16 bytes from global vaddr into LDS at vdst, tracked by ASYNCcnt.
// LDS flat address low 32 bits == LDS byte offset (ISA aperture rule).
__device__ __forceinline__ void async_g2l_b128(void* lds_dst, const void* gsrc) {
  unsigned ldsaddr = (unsigned)(unsigned long long)lds_dst;
  unsigned long long ga = (unsigned long long)gsrc;
  asm volatile("global_load_async_to_lds_b128 %0, %1, off"
               :: "v"(ldsaddr), "v"(ga)
               : "memory");
}
__device__ __forceinline__ void wait_async0() {
  asm volatile("s_wait_asynccnt 0" ::: "memory");
}

// ---------------------------------------------------------------------------
// Generic WMMA GEMM:  C[m,n] = scale * sum_k A[m,k] * W(k,n)  (+ bias[n])
//   A: fp32 or bf16 (A_BF16), row-major, leading dim lda (bf16 A: 16B aligned
//      rows, lda multiple of 8 -> 128-bit fragment loads)
//   W: fp32; B_KMAJOR==0 -> stored (N,K) row-major, else (K,N) row-major
//   C: fp32 or bf16 (C_BF16), leading dim ldc
// Block = 128 threads = 4 waves in a 2x2 grid; each wave owns a 32x32 tile
// (2x2 WMMA fragments with A/B reuse -> 4 WMMA per K-step per wave).
// Block tile = 64x64.  All M,N used are multiples of 64, K multiples of 32.
// ---------------------------------------------------------------------------
template<int A_BF16, int B_KMAJOR, int C_BF16>
__global__ void __launch_bounds__(128)
wmma_gemm_kernel(const void* __restrict__ Aptr, long lda,
                 const float* __restrict__ Bw, long ldb,
                 void* __restrict__ Cptr, long ldc,
                 const float* __restrict__ bias, float out_scale, int K) {
  const int tid  = threadIdx.x;
  const int wv   = tid >> 5;
  const int lane = tid & 31;
  const int l    = lane & 15;
  const int hi   = lane >> 4;
  const int mw   = wv >> 1;
  const int nw   = wv & 1;
  const long m0  = (long)blockIdx.y * 64 + mw * 32;
  const long n0  = (long)blockIdx.x * 64 + nw * 32;

  const v8f zero = {0.f,0.f,0.f,0.f,0.f,0.f,0.f,0.f};
  v8f acc[2][2];
  acc[0][0] = zero; acc[0][1] = zero; acc[1][0] = zero; acc[1][1] = zero;

  const float*  Af = (const float*)Aptr;
  const __bf16* Ah = (const __bf16*)Aptr;

  for (int kb = 0; kb < K; kb += 32) {
    Frag16 af[2], bf[2];
    // A fragments 16x32: lanes 0-15 K{0-15}, lanes 16-31 K{16-31}
    #pragma unroll
    for (int mi = 0; mi < 2; mi++) {
      const long aoff = (m0 + 16 * mi + l) * lda + kb + (hi ? 16 : 0);
      if (A_BF16) {
        const v4u* p = (const v4u*)(Ah + aoff);   // 16B aligned
        af[mi].u4[0] = p[0];                      // K elems +0..7
        af[mi].u4[1] = p[1];                      // K elems +8..15
        if (kb + 32 < K) __builtin_prefetch(Ah + aoff + 32, 0, 1);
      } else {
        const float* p = Af + aoff;
        #pragma unroll
        for (int i = 0; i < 8; i++) {
          af[mi].h[i]     = (__bf16)p[i];
          af[mi].h[8 + i] = (__bf16)p[8 + i];
        }
        if (kb + 32 < K) __builtin_prefetch(p + 32, 0, 1);
      }
    }
    // B fragments 32x16: half-wave holds K halves, lane -> column
    #pragma unroll
    for (int ni = 0; ni < 2; ni++) {
      const long ncol = n0 + 16 * ni + l;
      if (!B_KMAJOR) {
        const float* p = Bw + ncol * ldb + kb + (hi ? 16 : 0);
        #pragma unroll
        for (int i = 0; i < 16; i++) bf[ni].h[i] = (__bf16)p[i];
        if (kb + 32 < K) __builtin_prefetch(p + 32, 0, 1);
      } else {
        const float* p = Bw + (long)(kb + (hi ? 16 : 0)) * ldb + ncol;
        #pragma unroll
        for (int i = 0; i < 16; i++) bf[ni].h[i] = (__bf16)p[(long)i * ldb];
      }
    }
    #pragma unroll
    for (int mi = 0; mi < 2; mi++)
      #pragma unroll
      for (int ni = 0; ni < 2; ni++)
        acc[mi][ni] = wmma_bf16(af[mi].v, bf[ni].v, acc[mi][ni]);
  }

  #pragma unroll
  for (int ni = 0; ni < 2; ni++) {
    const long col = n0 + 16 * ni + l;
    const float bv = bias ? bias[col] : 0.0f;
    #pragma unroll
    for (int mi = 0; mi < 2; mi++) {
      #pragma unroll
      for (int i = 0; i < 8; i++) {
        const long m = m0 + 16 * mi + (hi ? 8 : 0) + i;
        const float v = acc[mi][ni][i] * out_scale + bv;
        if (C_BF16) ((__bf16*)Cptr)[m * ldc + col] = (__bf16)v;
        else        ((float*)Cptr)[m * ldc + col] = v;
      }
    }
  }
}

// ---------------------------------------------------------------------------
// RMSNorm, one block (256 thr) per row, in-place fp32
// ---------------------------------------------------------------------------
__global__ void rmsnorm_inplace_kernel(float* __restrict__ x,
                                       const float* __restrict__ w, int width) {
  __shared__ float red[256];
  float* p = x + (long)blockIdx.x * width;
  float s = 0.f;
  for (int i = threadIdx.x; i < width; i += 256) { float v = p[i]; s += v * v; }
  red[threadIdx.x] = s; __syncthreads();
  for (int o = 128; o > 0; o >>= 1) {
    if (threadIdx.x < o) red[threadIdx.x] += red[threadIdx.x + o];
    __syncthreads();
  }
  const float r = rsqrtf(red[0] / (float)width + EPS_);
  for (int i = threadIdx.x; i < width; i += 256) p[i] = w[i] * p[i] * r;
}

// RMSNorm fp32 (ld_in) -> bf16 (ld_out), first `width` columns of each row
__global__ void rmsnorm_bf16_kernel(const float* __restrict__ x, long ld_in,
                                    const float* __restrict__ w,
                                    unsigned short* __restrict__ out, long ld_out,
                                    int width) {
  __shared__ float red[256];
  const float* p = x + (long)blockIdx.x * ld_in;
  __bf16* o = (__bf16*)out + (long)blockIdx.x * ld_out;
  float s = 0.f;
  for (int i = threadIdx.x; i < width; i += 256) { float v = p[i]; s += v * v; }
  red[threadIdx.x] = s; __syncthreads();
  for (int ofs = 128; ofs > 0; ofs >>= 1) {
    if (threadIdx.x < ofs) red[threadIdx.x] += red[threadIdx.x + ofs];
    __syncthreads();
  }
  const float r = rsqrtf(red[0] / (float)width + EPS_);
  for (int i = threadIdx.x; i < width; i += 256)
    o[i] = (__bf16)(w[i] * p[i] * r);
}

// ---------------------------------------------------------------------------
// RoPE for k_pe: kv_full[..., 512:576] (fp32) -> Kcat[..., 512:576] (bf16)
// ---------------------------------------------------------------------------
__global__ void rope_k_kernel(const float* __restrict__ kv_full,
                              const float* __restrict__ cosb,
                              const float* __restrict__ sinb,
                              unsigned short* __restrict__ kcat) {
  const int idx = blockIdx.x * blockDim.x + threadIdx.x;
  const int j   = idx & 31;
  const long row = idx >> 5;                 // b*S + s
  const int s = (int)(row % S_);
  const float* p = kv_full + row * (long)CAT_ + KVLORA_ + 2 * j;
  const float xr = p[0], xi = p[1];
  const float c = cosb[s * 32 + j], sn = sinb[s * 32 + j];
  __bf16* o = (__bf16*)kcat + row * (long)CAT_ + KVLORA_ + 2 * j;
  o[0] = (__bf16)(xr * c - xi * sn);
  o[1] = (__bf16)(xr * sn + xi * c);
}

// RoPE for q_pe (bf16 in q buffer) -> Qcat[..., 512:576], folded with SCALE
__global__ void rope_q_kernel(const unsigned short* __restrict__ qb,
                              const float* __restrict__ cosb,
                              const float* __restrict__ sinb,
                              unsigned short* __restrict__ qcat, float scale) {
  const int idx = blockIdx.x * blockDim.x + threadIdx.x;
  const int j   = idx & 31;
  const int h   = (idx >> 5) & (H_ - 1);
  const long row = idx >> 9;                 // b*S + s
  const int s = (int)(row % S_);
  const __bf16* p = (const __bf16*)qb + (row * H_ + h) * (long)QK_ + NOPE_ + 2 * j;
  const float xr = (float)p[0], xi = (float)p[1];
  const float c = cosb[s * 32 + j], sn = sinb[s * 32 + j];
  __bf16* o = (__bf16*)qcat + (row * H_ + h) * (long)CAT_ + KVLORA_ + 2 * j;
  o[0] = (__bf16)((xr * c - xi * sn) * scale);
  o[1] = (__bf16)((xr * sn + xi * c) * scale);
}

// ---------------------------------------------------------------------------
// Flash attention over the 512-dim latent space (absorbed MLA form).
// Grid: (S/32, H, B); block = 128 threads = 4 waves; 32 queries per block
// (2 sub-tiles of 16) so each K/V fragment feeds TWO WMMAs.
// Q/K tiles staged in LDS via async global->LDS b128 copies (ASYNCcnt).
// ---------------------------------------------------------------------------
__global__ void __launch_bounds__(128)
mla_attn_kernel(const unsigned short* __restrict__ Qcat,
                const unsigned short* __restrict__ Kcat,
                unsigned short* __restrict__ Olat) {
  const int b  = blockIdx.z;
  const int h  = blockIdx.y;
  const int s0 = blockIdx.x * 32;
  const int tid  = threadIdx.x;
  const int w    = tid >> 5;
  const int lane = tid & 31;
  const int l    = lane & 15;
  const int hi   = lane >> 4;

  __shared__ __align__(16) __bf16 sQ[32][CATP_];   // ~37.4 KB
  __shared__ __align__(16) __bf16 sK[64][CATP_];   // ~74.8 KB
  __shared__ __align__(16) float  sS[32][64];      // 8 KB
  __shared__ __align__(16) __bf16 sP[32][64];      // 4 KB
  __shared__ float wmaxS[4][32];
  __shared__ float rowMax[32], rowSum[32], corr[32], nmax[32];

  const long qRowStride = (long)H_ * CAT_;
  const __bf16* Qbase = (const __bf16*)Qcat +
                        (((long)b * S_ + s0) * H_ + h) * (long)CAT_;
  const __bf16* Kb    = (const __bf16*)Kcat + (long)b * S_ * CAT_;

  // stage Q tile once: 32 rows x 72 async 16-byte copies
  for (int ch = tid; ch < 32 * 72; ch += 128) {
    const int r = ch / 72, cj = (ch % 72) * 8;
    async_g2l_b128(&sQ[r][cj], Qbase + (long)r * qRowStride + cj);
  }
  if (tid < 32) { rowMax[tid] = -1e30f; rowSum[tid] = 0.0f; }
  wait_async0();
  __syncthreads();

  const v8f zero = {0.f,0.f,0.f,0.f,0.f,0.f,0.f,0.f};
  v8f acc[2][8];
  #pragma unroll
  for (int qt = 0; qt < 2; qt++)
    #pragma unroll
    for (int nt = 0; nt < 8; nt++) acc[qt][nt] = zero;

  for (int t0 = 0; t0 < S_; t0 += 64) {
    // ---- stage K tile: 64 rows x 72 async 16-byte copies (read once)
    for (int ch = tid; ch < 64 * 72; ch += 128) {
      const int key = ch / 72, cj = (ch % 72) * 8;
      async_g2l_b128(&sK[key][cj], Kb + (long)(t0 + key) * CAT_ + cj);
    }
    wait_async0();
    __syncthreads();

    // ---- scores: this wave's keys are t0 + 16*w + [0,16); two query tiles
    v8f sc[2]; sc[0] = zero; sc[1] = zero;
    #pragma unroll 2
    for (int kb = 0; kb < CAT_; kb += 32) {
      Frag16 bfr;
      const __bf16* kp = &sK[16 * w + l][kb + (hi ? 16 : 0)];
      bfr.u4[0] = *(const v4u*)(kp);
      bfr.u4[1] = *(const v4u*)(kp + 8);
      #pragma unroll
      for (int qt = 0; qt < 2; qt++) {
        Frag16 af;
        const __bf16* ap = &sQ[16 * qt + l][kb + (hi ? 16 : 0)];
        af.u4[0] = *(const v4u*)(ap);        // K elems +0..7
        af.u4[1] = *(const v4u*)(ap + 8);    // K elems +8..15
        sc[qt] = wmma_bf16(af.v, bfr.v, sc[qt]);
      }
    }

    // per-row max over this wave's 16 columns (lanes within each half-wave)
    #pragma unroll
    for (int qt = 0; qt < 2; qt++) {
      float pm[8];
      #pragma unroll
      for (int i = 0; i < 8; i++) pm[i] = sc[qt][i];
      #pragma unroll
      for (int off = 8; off >= 1; off >>= 1) {
        #pragma unroll
        for (int i = 0; i < 8; i++)
          pm[i] = fmaxf(pm[i], __shfl_xor(pm[i], off, 32));
      }
      #pragma unroll
      for (int i = 0; i < 8; i++)
        sS[16 * qt + (hi ? 8 : 0) + i][16 * w + l] = sc[qt][i];
      if (l == 0) {
        #pragma unroll
        for (int i = 0; i < 8; i++)
          wmaxS[w][16 * qt + (hi ? 8 : 0) + i] = pm[i];
      }
    }
    __syncthreads();

    if (tid < 32) {
      const int r = tid;
      const float mprev = rowMax[r];
      const float tm = fmaxf(fmaxf(wmaxS[0][r], wmaxS[1][r]),
                             fmaxf(wmaxS[2][r], wmaxS[3][r]));
      const float nm = fmaxf(mprev, tm);
      nmax[r]   = nm;
      corr[r]   = __expf(mprev - nm);
      rowMax[r] = nm;
    }
    __syncthreads();

    // exponentiate all 32x64 scores
    for (int j = tid; j < 32 * 64; j += 128) {
      const int r = j >> 6, k = j & 63;
      const float p = __expf(sS[r][k] - nmax[r]);
      sS[r][k] = p;
      sP[r][k] = (__bf16)p;
    }
    __syncthreads();

    if (tid < 32) {
      const int r = tid;
      float s = 0.f;
      #pragma unroll 8
      for (int k = 0; k < 64; k++) s += sS[r][k];
      rowSum[r] = rowSum[r] * corr[r] + s;
    }

    // rescale accumulators by corr[row]
    #pragma unroll
    for (int qt = 0; qt < 2; qt++) {
      float rf[8];
      #pragma unroll
      for (int i = 0; i < 8; i++) rf[i] = corr[16 * qt + (hi ? 8 : 0) + i];
      #pragma unroll
      for (int nt = 0; nt < 8; nt++)
        #pragma unroll
        for (int i = 0; i < 8; i++) acc[qt][nt][i] = acc[qt][nt][i] * rf[i];
    }

    // ---- P x V : this wave handles latent cols [128w, 128w+128)
    #pragma unroll
    for (int kk = 0; kk < 2; kk++) {
      const int kb2 = kk * 32;
      Frag16 pf[2];
      #pragma unroll
      for (int qt = 0; qt < 2; qt++) {
        const __bf16* pp = &sP[16 * qt + l][kb2 + (hi ? 16 : 0)];
        pf[qt].u4[0] = *(const v4u*)(pp);        // K elems +0..7
        pf[qt].u4[1] = *(const v4u*)(pp + 8);    // K elems +8..15
      }
      #pragma unroll
      for (int nt = 0; nt < 8; nt++) {
        Frag16 vf;
        const int c = 128 * w + 16 * nt + l;
        #pragma unroll
        for (int i = 0; i < 16; i++)
          vf.h[i] = sK[kb2 + (hi ? 16 : 0) + i][c];
        #pragma unroll
        for (int qt = 0; qt < 2; qt++)
          acc[qt][nt] = wmma_bf16(pf[qt].v, vf.v, acc[qt][nt]);
      }
    }
    __syncthreads();
  }

  #pragma unroll
  for (int qt = 0; qt < 2; qt++) {
    float inv[8];
    #pragma unroll
    for (int i = 0; i < 8; i++)
      inv[i] = 1.0f / rowSum[16 * qt + (hi ? 8 : 0) + i];
    #pragma unroll
    for (int nt = 0; nt < 8; nt++) {
      const int c = 128 * w + 16 * nt + l;
      #pragma unroll
      for (int i = 0; i < 8; i++) {
        const long r = s0 + 16 * qt + (hi ? 8 : 0) + i;
        ((__bf16*)Olat)[(((long)b * S_ + r) * H_ + h) * (long)KVLORA_ + c] =
            (__bf16)(acc[qt][nt][i] * inv[i]);
      }
    }
  }
}

// ---------------------------------------------------------------------------
static void launch_gemm(const void* A, int a_bf16, long lda,
                        const float* Bw, int b_kmajor, long ldb,
                        void* C, int c_bf16, long ldc,
                        const float* bias, float scale,
                        int M, int N, int K, hipStream_t st) {
  dim3 grid(N / 64, M / 64);
  if (!a_bf16 && !b_kmajor && !c_bf16)
    wmma_gemm_kernel<0,0,0><<<grid, 128, 0, st>>>(A, lda, Bw, ldb, C, ldc, bias, scale, K);
  else if (!a_bf16 && !b_kmajor && c_bf16)
    wmma_gemm_kernel<0,0,1><<<grid, 128, 0, st>>>(A, lda, Bw, ldb, C, ldc, bias, scale, K);
  else if (a_bf16 && b_kmajor && c_bf16)
    wmma_gemm_kernel<1,1,1><<<grid, 128, 0, st>>>(A, lda, Bw, ldb, C, ldc, bias, scale, K);
  else if (a_bf16 && !b_kmajor && c_bf16)
    wmma_gemm_kernel<1,0,1><<<grid, 128, 0, st>>>(A, lda, Bw, ldb, C, ldc, bias, scale, K);
  else
    wmma_gemm_kernel<1,0,0><<<grid, 128, 0, st>>>(A, lda, Bw, ldb, C, ldc, bias, scale, K);
}

extern "C" void kernel_launch(void* const* d_in, const int* in_sizes, int n_in,
                              void* d_out, int out_size, void* d_ws, size_t ws_size,
                              hipStream_t stream) {
  (void)in_sizes; (void)n_in; (void)out_size; (void)ws_size;
  const float* x        = (const float*)d_in[0];
  /* d_in[1] = start_pos == 0, unused */
  const float* cosb     = (const float*)d_in[2];
  const float* sinb     = (const float*)d_in[3];
  const float* wq_a_w   = (const float*)d_in[4];
  const float* wq_a_b   = (const float*)d_in[5];
  const float* q_norm_w = (const float*)d_in[6];
  const float* wq_b_w   = (const float*)d_in[7];
  const float* wq_b_b   = (const float*)d_in[8];
  const float* wkv_a_w  = (const float*)d_in[9];
  const float* wkv_a_b  = (const float*)d_in[10];
  const float* kv_norm_w= (const float*)d_in[11];
  const float* wkv_b_w  = (const float*)d_in[12];
  const float* wo_w     = (const float*)d_in[13];
  const float* wo_b     = (const float*)d_in[14];
  float* out = (float*)d_out;

  const int M = B_ * S_;                       // 4096 token rows
  const float SCALE = 0.07216878364870323f;    // (NOPE+ROPE)^-0.5 = 192^-0.5

  // ---- workspace carve (256B aligned) ----
  char* ws = (char*)d_ws;
  size_t off = 0;
  auto carve = [&](size_t bytes) {
    void* p = ws + off;
    off = (off + bytes + 255) & ~(size_t)255;
    return p;
  };
  float*          q_lat   = (float*)carve((size_t)M * QLORA_ * 4);      // 16 MB
  float*          kv_full = (float*)carve((size_t)M * CAT_ * 4);        //  9 MB
  unsigned short* q_bf    = (unsigned short*)carve((size_t)M * H_ * QK_ * 2);     // 24 MB
  unsigned short* Qcat    = (unsigned short*)carve((size_t)M * H_ * CAT_ * 2);    // 72 MB
  unsigned short* Kcat    = (unsigned short*)carve((size_t)M * CAT_ * 2);         // 4.5 MB
  unsigned short* o_lat   = (unsigned short*)carve((size_t)M * H_ * KVLORA_ * 2); // 64 MB
  unsigned short* o_small = (unsigned short*)carve((size_t)M * (H_ * VH_) * 2);   // 16 MB

  // 1) q_lat = x @ wq_a^T + b
  launch_gemm(x, 0, D_, wq_a_w, 0, D_, q_lat, 0, QLORA_, wq_a_b, 1.0f,
              M, QLORA_, D_, stream);
  // 2) rms_norm(q_lat) in place
  rmsnorm_inplace_kernel<<<M, 256, 0, stream>>>(q_lat, q_norm_w, QLORA_);
  // 3) q = q_lat @ wq_b^T + b  -> bf16 (B,S,H,QK)
  launch_gemm(q_lat, 0, QLORA_, wq_b_w, 0, QLORA_, q_bf, 1, H_ * QK_,
              wq_b_b, 1.0f, M, H_ * QK_, QLORA_, stream);
  // 4) kv_full = x @ wkv_a^T + b
  launch_gemm(x, 0, D_, wkv_a_w, 0, D_, kv_full, 0, CAT_, wkv_a_b, 1.0f,
              M, CAT_, D_, stream);
  // 5) kv_c = rms_norm(kv_full[:,:512]) -> Kcat[:, 0:512] bf16
  rmsnorm_bf16_kernel<<<M, 256, 0, stream>>>(kv_full, CAT_, kv_norm_w,
                                             Kcat, CAT_, KVLORA_);
  // 6) rope(k_pe) -> Kcat[:, 512:576]
  rope_k_kernel<<<(M * 32) / 256, 256, 0, stream>>>(kv_full, cosb, sinb, Kcat);
  // 7) rope(q_pe) * SCALE -> Qcat[..., 512:576]
  rope_q_kernel<<<(M * H_ * 32) / 256, 256, 0, stream>>>(q_bf, cosb, sinb,
                                                         Qcat, SCALE);
  // 8) absorbed q: Qcat[...,0:512] = SCALE * q_nope @ wkv_b[h,:128,:]   (per head)
  for (int h = 0; h < H_; h++) {
    launch_gemm(q_bf + (size_t)h * QK_, 1, (long)H_ * QK_,
                wkv_b_w + (size_t)h * (NOPE_ + VH_) * KVLORA_, 1, KVLORA_,
                Qcat + (size_t)h * CAT_, 1, (long)H_ * CAT_,
                nullptr, SCALE, M, KVLORA_, NOPE_, stream);
  }
  // 9) flash attention in latent space -> o_lat bf16 (B,S,H,512)
  mla_attn_kernel<<<dim3(S_ / 32, H_, B_), 128, 0, stream>>>(Qcat, Kcat, o_lat);
  // 10) absorbed v: o_small[...,h,:] = o_lat[h] @ wkv_b[h,128:,:]^T    (per head)
  for (int h = 0; h < H_; h++) {
    launch_gemm(o_lat + (size_t)h * KVLORA_, 1, (long)H_ * KVLORA_,
                wkv_b_w + ((size_t)h * (NOPE_ + VH_) + NOPE_) * KVLORA_, 0, KVLORA_,
                o_small + (size_t)h * VH_, 1, (long)H_ * VH_,
                nullptr, 1.0f, M, VH_, KVLORA_, stream);
  }
  // 11) out = o_small @ wo^T + b  (fp32)
  launch_gemm(o_small, 1, (long)H_ * VH_, wo_w, 0, H_ * VH_,
              out, 0, D_, wo_b, 1.0f, M, D_, H_ * VH_, stream);
}
